// PFNLayerV9_44092134261312
// MI455X (gfx1250) — matrix operand order repaired and checked by reading, not compiled, not run on hardware
//
#include <hip/hip_runtime.h>
#include <hip/hip_bf16.h>

typedef __attribute__((ext_vector_type(16))) _Float16 v16h;
typedef __attribute__((ext_vector_type(8)))  _Float16 v8h;
typedef __attribute__((ext_vector_type(8)))  float    v8f;

#define N_PTS   1048576
#define CH      64
#define NSEG    65536

// ---------------- workspace layout (bytes) ----------------
// esum   : float [NSEG*64]           @ 0           (16 MB)
// wmaxE  : uint  [NSEG*64]           @ 16777216    (16 MB)
// cnt    : float [NSEG]              @ 33554432    (256 KB)
// packs  : _Float16 [3*4096]         @ 33816576    (24 KB)   (Wkv, W1, W2)
// stats  : float [256]               @ 33841152    (sumV,sumV2,sumH,sumH2)
// coef   : float [256]               @ 33842176    (aV,cV,aH,cH)
static constexpr size_t OFF_ESUM  = 0;
static constexpr size_t OFF_WMAX  = 16777216;
static constexpr size_t OFF_CNT   = 33554432;
static constexpr size_t OFF_PACK  = 33816576;
static constexpr size_t OFF_STAT  = 33841152;
static constexpr size_t OFF_COEF  = 33842176;

// ---------------- WMMA tile loaders ----------------
// A (16x32 f16, ISA 7.12.2 dense table): lanes 0-15 row M=lane hold K = k0+{0..7, 16..23},
// lanes 16-31 row M=lane-16 hold K = k0+{8..15, 24..31}.
__device__ __forceinline__ v8h cvt8(v8f x) {
  v8h r;
#pragma unroll
  for (int i = 0; i < 8; ++i) r[i] = (_Float16)x[i];
  return r;
}

__device__ __forceinline__ v16h loadA_f32(const float* __restrict__ X, int r0, int k0, int lane) {
  int m = lane & 15, half = (lane >> 4) & 1;
  const float* row = X + (size_t)(r0 + m) * CH;
  v16h a;
  ((v8h*)&a)[0] = cvt8(*(const v8f*)(row + k0 + 8 * half));
  ((v8h*)&a)[1] = cvt8(*(const v8f*)(row + k0 + 16 + 8 * half));
  return a;
}

__device__ __forceinline__ v16h loadA_lds(const _Float16* __restrict__ tile /*16 x 72*/, int k0, int lane) {
  int m = lane & 15, half = (lane >> 4) & 1;
  const _Float16* row = tile + m * 72;
  v16h a;
  ((v8h*)&a)[0] = *(const v8h*)(row + k0 + 8 * half);
  ((v8h*)&a)[1] = *(const v8h*)(row + k0 + 16 + 8 * half);
  return a;
}

// B packed buffer: p = tile*1024 + chunk*512 + lane*16 + j ; element = W[k][n],
// n = tile*16 + (lane&15), k = chunk*32 + 8*(lane>>4) + (j<8 ? j : j+8)   (B 32x16 striping).
__device__ __forceinline__ v16h loadB(const _Float16* __restrict__ pk, int tile, int chunk, int lane) {
  const v8h* p = (const v8h*)(pk + tile * 1024 + chunk * 512 + lane * 16);
  v16h b;
  ((v8h*)&b)[0] = p[0];
  ((v8h*)&b)[1] = p[1];
  return b;
}

// full 16x64 (K=64) product: 8 x v_wmma_f32_16x16x32_f16
__device__ __forceinline__ void gemm64(const v16h& a0, const v16h& a1,
                                       const _Float16* __restrict__ pk, int lane, v8f acc[4]) {
#pragma unroll
  for (int t = 0; t < 4; ++t) {
    v8f c = {0.f, 0.f, 0.f, 0.f, 0.f, 0.f, 0.f, 0.f};
    v16h b0 = loadB(pk, t, 0, lane);
    v16h b1 = loadB(pk, t, 1, lane);
    c = __builtin_amdgcn_wmma_f32_16x16x32_f16(false, a0, false, b0, (short)0, c, false, false);
    c = __builtin_amdgcn_wmma_f32_16x16x32_f16(false, a1, false, b1, (short)0, c, false, false);
    acc[t] = c;
  }
}

// float -> order-preserving uint (umax == fmax for non-NaN)
__device__ __forceinline__ unsigned encf(float f) {
  unsigned u = __float_as_uint(f);
  return u ^ ((u & 0x80000000u) ? 0xFFFFFFFFu : 0x80000000u);
}
__device__ __forceinline__ float decf(unsigned u) {
  return __uint_as_float((u & 0x80000000u) ? (u ^ 0x80000000u) : ~u);
}

// ---------------- kernels ----------------
__global__ __launch_bounds__(256) void k_init(float* esum, unsigned* wmaxE, float* cnt,
                                              float* gstats, float* out) {
  int i = blockIdx.x * 256 + threadIdx.x;           // < NSEG*64
  esum[i] = 0.f;
  wmaxE[i] = 0x007FFFFFu;                           // enc(-inf)
  int s = i >> 6, n = i & 63;
  out[(size_t)s * 128 + n] = 0.f;                   // evsum accumulator
  ((unsigned*)out)[(size_t)s * 128 + 64 + n] = 0x80000000u;  // enc(0) for q_max (x>=0)
  if (i < NSEG) cnt[i] = 0.f;
  if (i < 256) gstats[i] = 0.f;
}

__global__ __launch_bounds__(256) void k_pack(const float* __restrict__ Wkv,
                                              const float* __restrict__ W1,
                                              const float* __restrict__ W2,
                                              _Float16* __restrict__ packs) {
  int idx = blockIdx.x * 256 + threadIdx.x;         // < 3*4096
  int m = idx >> 12;
  int p = idx & 4095;
  const float* W = (m == 0) ? Wkv : (m == 1) ? W1 : W2;
  int j = p & 15, lane = (p >> 4) & 31, chunk = (p >> 9) & 1, tile = p >> 10;
  int n = tile * 16 + (lane & 15);
  int k = chunk * 32 + ((j < 8) ? 8 * (lane >> 4) + j : 16 + 8 * (lane >> 4) + (j - 8));
  packs[idx] = (_Float16)W[k * CH + n];             // W row-major [in][out]
}

__global__ __launch_bounds__(256) void k_stats(const float* __restrict__ X,
                                               const _Float16* __restrict__ pkKV,
                                               const _Float16* __restrict__ pk1,
                                               float* __restrict__ gstats) {
  __shared__ float ls[256];
  int tid = threadIdx.x;
  ls[tid] = 0.f;
  __syncthreads();
  int lane = tid & 31, wave = tid >> 5;
  int r0 = blockIdx.x * 128 + wave * 16;
  v16h a0 = loadA_f32(X, r0, 0, lane);
  v16h a1 = loadA_f32(X, r0, 32, lane);
  v8f av[4], ah[4];
  gemm64(a0, a1, pkKV, lane, av);
  gemm64(a0, a1, pk1, lane, ah);
#pragma unroll
  for (int t = 0; t < 4; ++t) {
    int n = t * 16 + (lane & 15);
    float s = 0.f, s2 = 0.f, sh = 0.f, sh2 = 0.f;
#pragma unroll
    for (int j = 0; j < 8; ++j) {
      float v = av[t][j], h = ah[t][j];
      s += v;  s2 += v * v;  sh += h;  sh2 += h * h;
    }
    atomicAdd(&ls[n], s);
    atomicAdd(&ls[64 + n], s2);
    atomicAdd(&ls[128 + n], sh);
    atomicAdd(&ls[192 + n], sh2);
  }
  __syncthreads();
  atomicAdd(&gstats[tid], ls[tid]);
}

__global__ __launch_bounds__(64) void k_coef(const float* __restrict__ gstats,
                                             const float* __restrict__ gnorm,
                                             const float* __restrict__ bnorm,
                                             const float* __restrict__ g1,
                                             const float* __restrict__ beta1,
                                             float* __restrict__ coef) {
  int n = threadIdx.x;                              // 64
  const float inv = 1.f / (float)N_PTS;
  float mV = gstats[n] * inv;
  float vV = gstats[64 + n] * inv - mV * mV;
  float aV = gnorm[n] * rsqrtf(vV + 1e-3f);
  float mH = gstats[128 + n] * inv;
  float vH = gstats[192 + n] * inv - mH * mH;
  float aH = g1[n] * rsqrtf(vH + 1e-5f);
  coef[n]        = aV;
  coef[64 + n]   = bnorm[n] - aV * mV;
  coef[128 + n]  = aH;
  coef[192 + n]  = beta1[n] - aH * mH;
}

__global__ __launch_bounds__(256) void k_pass2(const float* __restrict__ X,
                                               const int* __restrict__ seg,
                                               const _Float16* __restrict__ pkKV,
                                               const _Float16* __restrict__ pk1,
                                               const _Float16* __restrict__ pk2,
                                               const float* __restrict__ coef,
                                               unsigned* __restrict__ wmaxE,
                                               float* __restrict__ cnt,
                                               float* __restrict__ out) {
  __shared__ int sseg[128];
  __shared__ _Float16 hT[8][16 * 72];
  int tid = threadIdx.x, lane = tid & 31, wave = tid >> 5;
  int rB = blockIdx.x * 128;
  if (tid < 128) {
    int s = seg[rB + tid];
    sseg[tid] = s;
    atomicAdd(&cnt[s], 1.f);
  }
  __syncthreads();
  int r0 = rB + wave * 16;
  v16h a0 = loadA_f32(X, r0, 0, lane);
  v16h a1 = loadA_f32(X, r0, 32, lane);
  v8f av[4], ah[4];
  gemm64(a0, a1, pkKV, lane, av);
  gemm64(a0, a1, pk1, lane, ah);
  int nl = lane & 15, half = (lane >> 4) & 1;
  unsigned* qmaxE = (unsigned*)out;
#pragma unroll
  for (int t = 0; t < 4; ++t) {
    int n = t * 16 + nl;
    float aV = coef[n], cV = coef[64 + n], aH = coef[128 + n], cH = coef[192 + n];
#pragma unroll
    for (int j = 0; j < 8; ++j) {
      int row = wave * 16 + 8 * half + j;
      int s = sseg[row];
      float x = fmaxf(av[t][j] * aV + cV, 0.f);     // relu(bn(v))
      atomicMax(&qmaxE[(size_t)s * 128 + 64 + n], __float_as_uint(x) | 0x80000000u);
      float h = fmaxf(ah[t][j] * aH + cH, 0.f);     // relu(bn(h_pre))
      hT[wave][(8 * half + j) * 72 + n] = (_Float16)h;
    }
  }
  __syncthreads();
  v16h h0 = loadA_lds(&hT[wave][0], 0, lane);
  v16h h1 = loadA_lds(&hT[wave][0], 32, lane);
  v8f aw[4];
  gemm64(h0, h1, pk2, lane, aw);                    // weight (bias b2 cancels in softmax)
#pragma unroll
  for (int t = 0; t < 4; ++t) {
    int n = t * 16 + nl;
#pragma unroll
    for (int j = 0; j < 8; ++j) {
      int row = wave * 16 + 8 * half + j;
      int s = sseg[row];
      atomicMax(&wmaxE[(size_t)s * 64 + n], encf(aw[t][j]));
    }
  }
}

__global__ __launch_bounds__(256) void k_pass3(const float* __restrict__ X,
                                               const int* __restrict__ seg,
                                               const _Float16* __restrict__ pkKV,
                                               const _Float16* __restrict__ pk1,
                                               const _Float16* __restrict__ pk2,
                                               const float* __restrict__ coef,
                                               const float* __restrict__ bkv,
                                               const unsigned* __restrict__ wmaxE,
                                               float* __restrict__ esum,
                                               float* __restrict__ out) {
  __shared__ int sseg[128];
  __shared__ _Float16 hT[8][16 * 72];
  int tid = threadIdx.x, lane = tid & 31, wave = tid >> 5;
  int rB = blockIdx.x * 128;
  if (tid < 128) sseg[tid] = seg[rB + tid];
  __syncthreads();
  int r0 = rB + wave * 16;
  v16h a0 = loadA_f32(X, r0, 0, lane);
  v16h a1 = loadA_f32(X, r0, 32, lane);
  v8f av[4], ah[4];
  gemm64(a0, a1, pkKV, lane, av);
  gemm64(a0, a1, pk1, lane, ah);
  int nl = lane & 15, half = (lane >> 4) & 1;
#pragma unroll
  for (int t = 0; t < 4; ++t) {
    int n = t * 16 + nl;
    float aH = coef[128 + n], cH = coef[192 + n];
#pragma unroll
    for (int j = 0; j < 8; ++j) {
      float h = fmaxf(ah[t][j] * aH + cH, 0.f);
      hT[wave][(8 * half + j) * 72 + n] = (_Float16)h;
    }
  }
  __syncthreads();
  v16h h0 = loadA_lds(&hT[wave][0], 0, lane);
  v16h h1 = loadA_lds(&hT[wave][0], 32, lane);
  v8f aw[4];
  gemm64(h0, h1, pk2, lane, aw);
#pragma unroll
  for (int t = 0; t < 4; ++t) {
    int n = t * 16 + nl;
    float bk = bkv[n];
#pragma unroll
    for (int j = 0; j < 8; ++j) {
      int row = wave * 16 + 8 * half + j;
      int s = sseg[row];
      float wm = decf(wmaxE[(size_t)s * 64 + n]);
      float e = __expf(aw[t][j] - wm);
      atomicAdd(&esum[(size_t)s * 64 + n], e);
      atomicAdd(&out[(size_t)s * 128 + n], e * (av[t][j] + bk));  // v includes b_kv
    }
  }
}

__global__ __launch_bounds__(256) void k_final(const float* __restrict__ esum,
                                               const float* __restrict__ cnt,
                                               float* __restrict__ out) {
  int i = blockIdx.x * 256 + threadIdx.x;           // < NSEG*64
  int s = i >> 6, n = i & 63;
  float es = esum[i];
  float ev = out[(size_t)s * 128 + n];
  float c = cnt[s];
  out[(size_t)s * 128 + n] = (es > 0.f) ? (ev / es) / fmaxf(c, 1.f) : 0.f;
  unsigned q = ((unsigned*)out)[(size_t)s * 128 + 64 + n];
  out[(size_t)s * 128 + 64 + n] = __uint_as_float(q & 0x7FFFFFFFu);  // decode (x>=0; empty -> 0)
}

extern "C" void kernel_launch(void* const* d_in, const int* in_sizes, int n_in,
                              void* d_out, int out_size, void* d_ws, size_t ws_size,
                              hipStream_t stream) {
  const float* X     = (const float*)d_in[0];
  const int*   segp  = (const int*)d_in[1];
  const float* Wkv   = (const float*)d_in[2];
  const float* bkv   = (const float*)d_in[3];
  const float* W1    = (const float*)d_in[4];
  const float* g1    = (const float*)d_in[6];
  const float* beta1 = (const float*)d_in[7];
  const float* W2    = (const float*)d_in[8];
  const float* gnorm = (const float*)d_in[10];
  const float* bnorm = (const float*)d_in[11];
  float* out = (float*)d_out;

  char* ws = (char*)d_ws;
  float*    esum   = (float*)(ws + OFF_ESUM);
  unsigned* wmaxE  = (unsigned*)(ws + OFF_WMAX);
  float*    cnt    = (float*)(ws + OFF_CNT);
  _Float16* packs  = (_Float16*)(ws + OFF_PACK);
  float*    gstats = (float*)(ws + OFF_STAT);
  float*    coef   = (float*)(ws + OFF_COEF);
  _Float16* pkKV = packs;
  _Float16* pk1  = packs + 4096;
  _Float16* pk2  = packs + 8192;

  k_init  <<<(NSEG * 64) / 256, 256, 0, stream>>>(esum, wmaxE, cnt, gstats, out);
  k_pack  <<<(3 * 4096) / 256, 256, 0, stream>>>(Wkv, W1, W2, packs);
  k_stats <<<N_PTS / 128, 256, 0, stream>>>(X, pkKV, pk1, gstats);
  k_coef  <<<1, 64, 0, stream>>>(gstats, gnorm, bnorm, g1, beta1, coef);
  k_pass2 <<<N_PTS / 128, 256, 0, stream>>>(X, segp, pkKV, pk1, pk2, coef, wmaxE, cnt, out);
  k_pass3 <<<N_PTS / 128, 256, 0, stream>>>(X, segp, pkKV, pk1, pk2, coef, bkv, wmaxE, esum, out);
  k_final <<<(NSEG * 64) / 256, 256, 0, stream>>>(esum, cnt, out);
  (void)in_sizes; (void)n_in; (void)out_size; (void)ws_size;
}
